// ConvLSTM_34273839022588
// MI455X (gfx1250) — matrix operand order repaired and checked by reading, not compiled, not run on hardware
//
#include <hip/hip_runtime.h>
#include <hip/hip_bf16.h>
#include <cstddef>

// ---------------------------------------------------------------------------
// CDNA5 (gfx1250, wave32) ConvLSTM stack via implicit-GEMM WMMA.
// bf16 operands (v16bf), f32 accumulate (v8f) with v_wmma_f32_16x16x32_bf16.
//
// Row-GEMM formulation: K is split per kernel-row dy. Patch stored in LDS as
// [dy][xxl][CTP] with CTP = ceil((Cin+F)/8)*8 channel padding, which makes
// every per-lane A fragment two ALIGNED ds_load_b128 directly from the patch
// (no permute staging, no barriers in the K loop). Weights are pre-packed in
// exact WMMA fragment order so B fragments are two contiguous
// global_load_b128 per lane (L2-broadcast). Zero-padded weights absorb all
// K/N padding (finite activations * 0 = 0).
// ---------------------------------------------------------------------------

typedef __attribute__((ext_vector_type(16))) __bf16       v16bf;
typedef __attribute__((ext_vector_type(8)))  float        v8f;
typedef __attribute__((ext_vector_type(4)))  unsigned int v4u;

union FragU { v16bf v; v4u q[2]; };

__device__ __forceinline__ unsigned short f32_to_bf16(float f) {
  union { float f; unsigned int u; } v; v.f = f;
  unsigned int u = v.u;
  u += 0x7FFFu + ((u >> 16) & 1u);           // round-to-nearest-even
  return (unsigned short)(u >> 16);
}
__device__ __forceinline__ float hsig(float x) {
  return fminf(fmaxf(0.2f * x + 0.5f, 0.0f), 1.0f);
}

// f32 -> bf16 bulk convert (layer-0 input sequence)
__global__ void cvt_f32_bf16_kernel(const float* __restrict__ in,
                                    unsigned short* __restrict__ out, int n) {
  int i = blockIdx.x * blockDim.x + threadIdx.x;
  if (i < n) out[i] = f32_to_bf16(in[i]);
}

// Pack Wx[k,k,Cin,4F] + Wh[k,k,F,4F] into WMMA-B fragment order:
//   out[(((dy*CPR + kc)*NT + nt)*32 + lane)*16 + j]
//   n = nt*16 + (lane&15);  kappa = kc*32 + ((lane>=16)?8:0) + (j&7) + ((j>=8)?16:0)
//   kappa -> dx = kappa/CTP, c = kappa%CTP ; zero outside valid dx/c/n.
__global__ void pack_w_frag_kernel(const float* __restrict__ Wx,
                                   const float* __restrict__ Wh,
                                   unsigned short* __restrict__ out,
                                   int k, int Cin, int F, int NT, int CTP, int CPR) {
  int i = blockIdx.x * blockDim.x + threadIdx.x;
  int total = k * CPR * NT * 512;
  if (i >= total) return;
  int j    = i & 15;
  int lane = (i >> 4) & 31;
  int r    = i >> 9;                 // (dy*CPR + kc)*NT + nt
  int nt   = r % NT;
  int rc   = r / NT;                 // dy*CPR + kc
  int kc   = rc % CPR;
  int dy   = rc / CPR;
  int n    = (nt << 4) + (lane & 15);
  int kap  = (kc << 5) + ((lane >> 4) << 3) + (j & 7) + ((j >> 3) << 4);
  int dx   = kap / CTP;
  int c    = kap - dx * CTP;
  int CT   = Cin + F, F4 = 4 * F;
  float val = 0.0f;
  if (dx < k && c < CT && n < F4) {
    if (c < Cin) val = Wx[((dy * k + dx) * Cin + c) * F4 + n];
    else         val = Wh[((dy * k + dx) * F + (c - Cin)) * F4 + n];
  }
  out[i] = f32_to_bf16(val);
}

// ---------------------------------------------------------------------------
// One (layer, timestep) step. Block = 64 consecutive pixels of one image row
// x all 4F gates. 256 threads = 8 waves; each wave owns 1-2 16x16 WMMA tiles.
// ---------------------------------------------------------------------------
template <int KK, int CIN, int FC>
__global__ __launch_bounds__(256)
void convlstm_step_kernel(const unsigned short* __restrict__ xin,  // [B,T,H,W,CIN] bf16
                          const unsigned short* __restrict__ wpk,  // fragment-packed bf16
                          const float* __restrict__ bias,          // [4F]
                          float* __restrict__ cbuf,                // [B,H,W,F] f32 cell state
                          unsigned short* __restrict__ hseq,       // [B,T,H,W,F] bf16
                          float* __restrict__ outF,                // null or [B,T,H,W,F] f32
                          int t) {
  constexpr int T  = 10, H = 200, W = 200;
  constexpr int CT  = CIN + FC;
  constexpr int CTP = ((CT + 7) / 8) * 8;          // padded channels (16B align)
  constexpr int F4  = 4 * FC;
  constexpr int NP  = ((F4 + 15) / 16) * 16;       // 32 or 64
  constexpr int NT  = NP / 16;                     // 2 or 4 N-tiles
  constexpr int CPR = (KK * CTP + 31) / 32;        // K-chunks per kernel row
  constexpr int PB  = (KK - 1) / 2;                // TF SAME pad_before
  constexpr int TILES = 4 * NT;                    // 8 or 16
  constexpr int XT  = (W + 63) / 64;
  constexpr int HW  = H * W;
  constexpr int WP  = 64 + KK - 1;                 // patch width in pixels
  constexpr int RS  = WP * CTP;                    // patch row stride (halves)

  const int bx   = blockIdx.x;
  const int xt   = bx % XT;
  const int y    = (bx / XT) % H;
  const int bimg = bx / (XT * H);
  const int x0   = xt << 6;

  // Patch (+32 halves slack: last-chunk A reads may overrun; weights are 0 there)
  __shared__ __align__(16) unsigned short sP[KK * RS + 32];
  __shared__ float sZ[64 * NP];

  const int tid  = threadIdx.x;
  const int lane = tid & 31;
  const int wave = tid >> 5;

  // ---- load input patch ONCE: rows [y-PB, y-PB+KK) x cols [x0-PB, ..+WP) ----
  {
    const unsigned short* xb = xin + (size_t)((bimg * T + t) * HW) * CIN;
    const unsigned short* hb = hseq + (size_t)((bimg * T + (t - 1)) * HW) * FC;
    const bool hOK = (t > 0);
    constexpr int PE = KK * RS;
    for (int e = tid; e < PE; e += 256) {
      int dy  = e / RS;
      int r   = e - dy * RS;
      int xxl = r / CTP;
      int c   = r - xxl * CTP;                     // c < CTP; [CT,CTP) stays 0
      int yy  = y + dy - PB;
      int xx  = x0 + xxl - PB;
      unsigned short val = 0;
      if (c < CT && yy >= 0 && yy < H && xx >= 0 && xx < W) {
        int pix = yy * W + xx;
        if (c < CIN)  val = xb[pix * CIN + c];
        else if (hOK) val = hb[pix * FC + (c - CIN)];
      }
      sP[e] = val;
    }
    // zero the slack tail
    if (tid < 32) sP[KK * RS + tid] = 0;
  }

  v8f acc0 = {};
  v8f acc1 = {};

  // tile 0: tileId = wave; tile 1 (TILES==16 only): tileId = wave + 8
  constexpr int NTSH = (NT == 4) ? 2 : 1;
  const int mt0 = wave >> NTSH, nt0 = wave & (NT - 1);
  const int mt1 = (wave + 8) >> NTSH, nt1 = (wave + 8) & (NT - 1);

  // Per-lane base addresses (halves). A: kappa group at s, s+16 for j<8 / j>=8.
  const int sK = (lane >> 4) << 3;                           // 0 or 8
  const unsigned short* a0 = &sP[((mt0 << 4) + (lane & 15)) * CTP + sK];
  const unsigned short* a1 = &sP[((mt1 << 4) + (lane & 15)) * CTP + sK];
  const unsigned short* w0 = wpk + ((size_t)nt0 << 9) + (lane << 4);
  const unsigned short* w1 = wpk + ((size_t)nt1 << 9) + (lane << 4);

  __syncthreads();                                           // sP ready

  // ---- K loop: per kernel-row dy, CPR chunks; no barriers, no masking ----
  for (int dy = 0; dy < KK; ++dy) {
    const unsigned short* ar0 = a0 + dy * RS;
    const unsigned short* ar1 = a1 + dy * RS;
    const unsigned short* wr  = wpk ? (wpk + (size_t)dy * CPR * NT * 512) : wpk;
    #pragma unroll
    for (int kc = 0; kc < CPR; ++kc) {
      const int ka = kc << 5;                                // kappa base (halves)
      const size_t wo = (size_t)kc * (NT << 9);
      {
        FragU fa, fb;
        fa.q[0] = *(const v4u*)(ar0 + ka);
        fa.q[1] = *(const v4u*)(ar0 + ka + 16);
        fb.q[0] = *(const v4u*)(w0 + (wr - wpk) + wo);
        fb.q[1] = *(const v4u*)(w0 + (wr - wpk) + wo + 8);
        acc0 = __builtin_amdgcn_wmma_f32_16x16x32_bf16(false, fa.v, false, fb.v,
                                                       (short)0, acc0, false, false);
      }
      if constexpr (TILES == 16) {
        FragU fa, fb;
        fa.q[0] = *(const v4u*)(ar1 + ka);
        fa.q[1] = *(const v4u*)(ar1 + ka + 16);
        fb.q[0] = *(const v4u*)(w1 + (wr - wpk) + wo);
        fb.q[1] = *(const v4u*)(w1 + (wr - wpk) + wo + 8);
        acc1 = __builtin_amdgcn_wmma_f32_16x16x32_bf16(false, fa.v, false, fb.v,
                                                       (short)0, acc1, false, false);
      }
    }
    // pull next row's weight fragments toward L2/L0
    if (dy + 1 < KK)
      __builtin_prefetch(wpk + (size_t)(dy + 1) * CPR * NT * 512 + tid * 8, 0, 1);
  }

  // ---- spill z = GEMM + bias into LDS (C/D layout: VGPR v -> M=v(+8), N=lane%16)
  const int col = lane & 15;
  const int rh  = (lane >> 4) << 3;
  {
    int n4 = (nt0 << 4) + col;
    float bv = (n4 < F4) ? bias[n4] : 0.0f;
    #pragma unroll
    for (int v = 0; v < 8; ++v)
      sZ[((mt0 << 4) + rh + v) * NP + n4] = acc0[v] + bv;
  }
  if constexpr (TILES == 16) {
    int n4 = (nt1 << 4) + col;
    float bv = (n4 < F4) ? bias[n4] : 0.0f;
    #pragma unroll
    for (int v = 0; v < 8; ++v)
      sZ[((mt1 << 4) + rh + v) * NP + n4] = acc1[v] + bv;
  }
  __syncthreads();

  // ---- pointwise Keras LSTM update: gates [i,f,g,o] along 4F ----
  for (int e = tid; e < 64 * FC; e += 256) {
    int m  = e / FC;
    int cf = e - m * FC;
    int x  = x0 + m;
    if (x < W) {
      float zi = sZ[m * NP + cf];
      float zf = sZ[m * NP + FC + cf];
      float zg = sZ[m * NP + 2 * FC + cf];
      float zo = sZ[m * NP + 3 * FC + cf];
      int pix  = y * W + x;
      int cidx = (bimg * HW + pix) * FC + cf;
      float cp = (t > 0) ? cbuf[cidx] : 0.0f;
      float ig = hsig(zi), fg = hsig(zf), og = hsig(zo);
      float cn = fg * cp + ig * tanhf(zg);
      float h  = og * tanhf(cn);
      cbuf[cidx] = cn;
      int hidx = ((bimg * T + t) * HW + pix) * FC + cf;
      hseq[hidx] = f32_to_bf16(h);
      if (outF) outF[hidx] = h;
    }
  }
}

// ---------------------------------------------------------------------------
template <int KK, int CIN, int FC>
static void run_layer(const unsigned short* in, const unsigned short* wpk,
                      const float* bias, float* cbuf, unsigned short* out,
                      float* outF, hipStream_t stream) {
  const int grid = 4 * 200 * ((200 + 63) / 64);
  for (int t = 0; t < 10; ++t)
    convlstm_step_kernel<KK, CIN, FC><<<grid, 256, 0, stream>>>(
        in, wpk, bias, cbuf, out, outF, t);
}

extern "C" void kernel_launch(void* const* d_in, const int* in_sizes, int n_in,
                              void* d_out, int out_size, void* d_ws, size_t ws_size,
                              hipStream_t stream) {
  (void)in_sizes; (void)n_in; (void)out_size; (void)ws_size;

  const float* x = (const float*)d_in[0];
  const float* Wx[4] = {(const float*)d_in[1], (const float*)d_in[4],
                        (const float*)d_in[7], (const float*)d_in[10]};
  const float* Wh[4] = {(const float*)d_in[2], (const float*)d_in[5],
                        (const float*)d_in[8], (const float*)d_in[11]};
  const float* bb[4] = {(const float*)d_in[3], (const float*)d_in[6],
                        (const float*)d_in[9], (const float*)d_in[12]};

  const int Bn = 4, T = 10, H = 200, W = 200;
  const int kk_[4]  = {3, 5, 9, 12};
  const int Cin_[4] = {1, 8, 16, 16};
  const int Ff_[4]  = {8, 16, 16, 5};

  // --- carve workspace (~117 MB) ---
  char* wsp = (char*)d_ws;
  size_t off = 0;
  auto alloc = [&](size_t bytes) -> void* {
    void* p = wsp + off;
    off = (off + bytes + 255) & ~(size_t)255;
    return p;
  };
  const size_t seqMax = (size_t)Bn * T * H * W * 16 * 2;      // bf16, F<=16
  unsigned short* seqA  = (unsigned short*)alloc(seqMax);
  unsigned short* seqB  = (unsigned short*)alloc(seqMax);
  float*          cbuf  = (float*)alloc((size_t)Bn * H * W * 16 * 4);
  unsigned short* xseq0 = (unsigned short*)alloc((size_t)Bn * T * H * W * 2);

  int NT[4], CTP[4], CPR[4];
  unsigned short* wpk[4];
  for (int l = 0; l < 4; ++l) {
    int F4  = 4 * Ff_[l];
    NT[l]   = (((F4 + 15) / 16) * 16) / 16;                   // 2 or 4
    int CT  = Cin_[l] + Ff_[l];
    CTP[l]  = ((CT + 7) / 8) * 8;
    CPR[l]  = (kk_[l] * CTP[l] + 31) / 32;
    wpk[l]  = (unsigned short*)alloc((size_t)kk_[l] * CPR[l] * NT[l] * 512 * 2);
  }

  // --- prep: input conversion + fragment-order weight packing ---
  {
    int n = Bn * T * H * W;                                   // Cin0 == 1
    cvt_f32_bf16_kernel<<<(n + 255) / 256, 256, 0, stream>>>(x, xseq0, n);
  }
  for (int l = 0; l < 4; ++l) {
    int total = kk_[l] * CPR[l] * NT[l] * 512;
    pack_w_frag_kernel<<<(total + 255) / 256, 256, 0, stream>>>(
        Wx[l], Wh[l], wpk[l], kk_[l], Cin_[l], Ff_[l], NT[l], CTP[l], CPR[l]);
  }

  // --- 4 layers x T timesteps, ping-pong sequence buffers ---
  run_layer<3, 1, 8>  (xseq0, wpk[0], bb[0], cbuf, seqA, nullptr,        stream);
  run_layer<5, 8, 16> (seqA,  wpk[1], bb[1], cbuf, seqB, nullptr,        stream);
  run_layer<9, 16, 16>(seqB,  wpk[2], bb[2], cbuf, seqA, nullptr,        stream);
  run_layer<12, 16, 5>(seqA,  wpk[3], bb[3], cbuf, seqB, (float*)d_out,  stream);
}
